// HoltWintersNoTrend_24343874633908
// MI455X (gfx1250) — matrix operand order repaired and checked by reading, not compiled, not run on hardware
//
#include <hip/hip_runtime.h>

// Holt-Winters (no trend) on gfx1250.
// Bandwidth-bound: 128 MB stream / 23.3 TB/s ~= 5.5 us floor.
// Sequential scan blocked into K=32 chunks executed as v_wmma_f32_16x16x32_f16
// with the carry folded into the C operand:
//   carry_new[m] = q^32*carry[m] + sum_k (alpha*q^(31-k)) * x[m][32c+k] - seasonSum[m, c%3]
// 16 series per wave (M dim); weight vector replicated across all 16 B columns.
// Season chunk-sums are periodic with period 3 in the chunk index (lcm(32,12)=96),
// and are computed ONCE, distributed across lanes (2 sums/lane) to keep setup tiny.

typedef __attribute__((ext_vector_type(16))) _Float16 v16h;
typedef __attribute__((ext_vector_type(8)))  float    v8f;

struct C4 { float4 a, b, c, d; };

__global__ __launch_bounds__(32)
void hw_scan_wmma(const float* __restrict__ series,
                  const float* __restrict__ alpha_p,
                  const float* __restrict__ season,
                  const int*   __restrict__ shifts,
                  float* __restrict__ out,
                  int L, int SLEN, int n_preds)
{
    const int ln   = threadIdx.x;        // 0..31 (wave32)
    const int base = blockIdx.x * 16;    // 16 series per wave
    const int g    = ln >> 4;            // WMMA lane half
    const int m    = ln & 15;            // row within 16-series tile

    const float alpha = alpha_p[0];
    const float q     = 1.0f - alpha;
    const float rq    = 1.0f / q;

    // powers of q
    float qp = 1.0f, q15 = 1.0f;
#pragma clang loop unroll(disable)
    for (int i = 0; i < 31; ++i) { qp *= q; if (i == 14) q15 = qp; }
    const float q31 = qp;
    const float d32 = q31 * q;

    // season values live in lanes 0..SLEN-1; fetched via ds_bpermute
    const float seasonB = season[ln < SLEN ? ln : 0];
    auto sfl = [](float v, int l) { return __shfl(v, l, 32); };

    const float* __restrict__ row = series + (size_t)(base + m) * L;
    const int shv = shifts[base + m];

    // ---- chunk 0 (t = 0..31) scalar, exact f32 (lanes 16..31 duplicate) ----
    float smooth = row[0];
    {
        int idx = (13 - shv) % 12;                 // phase of t=1
#pragma clang loop unroll(disable)
        for (int t = 1; t < 32; ++t) {
            float sv = sfl(seasonB, idx);
            smooth = alpha * (row[t] - sv) + q * smooth;
            idx = (idx + 1 == 12) ? 0 : idx + 1;
        }
    }

    // carry in C/D layout: carry[r] holds row (r + 8*g)
    v8f carry;
#pragma unroll
    for (int r = 0; r < 8; ++r) {
        float lo = sfl(smooth, r);
        float hi = sfl(smooth, r + 8);
        carry[r] = g ? hi : lo;
    }

    // ---- B fragment: w[k] = alpha*q^(31-k), replicated in all 16 columns ----
    // B K-order chosen here (same as A): lanes 0-15 hold K=0..15, lanes 16-31 hold K=16..31
    v16h bfrag;
    {
        float wc = alpha * (g ? q15 : q31);
#pragma unroll
        for (int e = 0; e < 16; ++e) { bfrag[e] = (_Float16)wc; wc *= rq; }
    }

    // ---- season chunk sums, DISTRIBUTED: each lane computes 2 of the 48 sums ----
    // S(row,p) = sum_k w[k]*season[(o_p + k - sh_row) mod 12], o_p in {0,8,4}.
    // accA(lane l) = S(l&15, l>>4)  (offset g?8:0);  accB(lane l) = S(l&15, 2) (offset 4).
    float accA = 0.0f, accB = 0.0f;
    {
        int idxA = ((g ? 8 : 0) + 12 - shv) % 12;
        int idxB = (4 + 12 - shv) % 12;
        float wk = alpha * q31;
#pragma clang loop unroll(disable)
        for (int k = 0; k < 32; ++k) {
            accA += wk * sfl(seasonB, idxA);
            accB += wk * sfl(seasonB, idxB);
            wk *= rq;
            idxA = (idxA + 1 == 12) ? 0 : idxA + 1;
            idxB = (idxB + 1 == 12) ? 0 : idxB + 1;
        }
    }
    // Gather into C/D layout: lane needs S(r+8g, p)
    float s3_0[8], s3_1[8], s3_2[8];
#pragma unroll
    for (int r = 0; r < 8; ++r) {
        s3_0[r] = sfl(accA, r + 8 * g);        // p=0 lives in lanes 0..15 of accA
        s3_1[r] = sfl(accA, 16 + r + 8 * g);   // p=1 lives in lanes 16..31 of accA
        s3_2[r] = sfl(accB, r + 8 * g);        // p=2 lives in accB (row = lane&15)
    }

    // ---- main WMMA scan over chunks c = 1 .. L/32-2 ----
    // A fragment K-order per lane: elems 0-15 -> K=16g..16g+15
    const int CN = L / 32 - 2;       // last WMMA chunk index (62 for L=2048)
    auto loadChunk = [&](int c) {
        const float4* rp = (const float4*)(row + 32 * c);
        C4 r4;
        r4.a = rp[4 * g];     r4.b = rp[4 * g + 1];
        r4.c = rp[4 * g + 2]; r4.d = rp[4 * g + 3];
        return r4;
    };
    auto toA = [](const C4& ch) {
        v16h a;
        a[0]  = (_Float16)ch.a.x; a[1]  = (_Float16)ch.a.y;
        a[2]  = (_Float16)ch.a.z; a[3]  = (_Float16)ch.a.w;
        a[4]  = (_Float16)ch.b.x; a[5]  = (_Float16)ch.b.y;
        a[6]  = (_Float16)ch.b.z; a[7]  = (_Float16)ch.b.w;
        a[8]  = (_Float16)ch.c.x; a[9]  = (_Float16)ch.c.y;
        a[10] = (_Float16)ch.c.z; a[11] = (_Float16)ch.c.w;
        a[12] = (_Float16)ch.d.x; a[13] = (_Float16)ch.d.y;
        a[14] = (_Float16)ch.d.z; a[15] = (_Float16)ch.d.w;
        return a;
    };

    C4 cur = loadChunk(1);
    auto step = [&](int c, const float (&s3)[8]) {
        C4 nxt = loadChunk(c + 1);                        // c=CN loads the tail chunk (in-bounds)
        int pc = c + 8; if (pc > L / 32 - 1) pc = L / 32 - 1;
        __builtin_prefetch(row + 32 * pc + 16 * g, 0, 1); // global_prefetch_b8
        v16h a = toA(cur);
        v8f cin;
#pragma unroll
        for (int r = 0; r < 8; ++r) cin[r] = d32 * carry[r] - s3[r];
        carry = __builtin_amdgcn_wmma_f32_16x16x32_f16(
            /*neg_a=*/false, a, /*neg_b=*/false, bfrag,
            /*c_mod=*/(short)0, cin, /*reuse_a=*/false, /*reuse_b=*/false);
        cur = nxt;
    };

    int c = 1;
#pragma clang loop unroll(disable)
    for (; c + 2 <= CN; c += 3) {        // c%3 pattern: 1,2,0 -> s3_1, s3_2, s3_0
        step(c,     s3_1);
        step(c + 1, s3_2);
        step(c + 2, s3_0);
    }
    if (c <= CN) { step(c, s3_1); ++c; }
    if (c <= CN) { step(c, s3_2); ++c; }

    // ---- extract this lane's row carry (row ln for lanes 0..15) ----
    float sm = 0.0f;
#pragma unroll
    for (int r = 0; r < 8; ++r) {
        float lo = sfl(carry[r], 0);     // row r      (from a group-0 lane)
        float hi = sfl(carry[r], 16);    // row r + 8  (from a group-1 lane)
        if (ln == r)     sm = lo;
        if (ln == r + 8) sm = hi;
    }

    // ---- tail: last 32 steps scalar f32, emit the final n_preds outputs ----
    {
        const int tstart = L - 32;
        int idx = ((tstart % 12) + 12 - shv) % 12;
        float* __restrict__ orow = out + (size_t)(base + m) * n_preds;
#pragma clang loop unroll(disable)
        for (int t = tstart; t < L; ++t) {
            float sv = sfl(seasonB, idx);
            float x  = row[t];
            sm = alpha * (x - sv) + q * sm;
            int oi = t - (L - n_preds);
            if (ln < 16 && oi >= 0) orow[oi] = sm + sv;
            idx = (idx + 1 == 12) ? 0 : idx + 1;
        }
    }
}

extern "C" void kernel_launch(void* const* d_in, const int* in_sizes, int n_in,
                              void* d_out, int out_size, void* d_ws, size_t ws_size,
                              hipStream_t stream) {
    const float* series  = (const float*)d_in[0];
    const float* alpha_p = (const float*)d_in[1];
    // d_in[2] = gamma (unused by the reference recurrence)
    const float* season  = (const float*)d_in[3];
    const int*   shifts  = (const int*)d_in[4];
    // d_in[5] = n_preds (device scalar; recover from out_size instead)

    const int B       = in_sizes[4];
    const int L       = in_sizes[0] / B;
    const int SLEN    = in_sizes[3];
    const int n_preds = out_size / B;

    dim3 grid(B / 16), block(32);        // one wave per block, 16 series per wave
    hw_scan_wmma<<<grid, block, 0, stream>>>(series, alpha_p, season, shifts,
                                             (float*)d_out, L, SLEN, n_preds);
}
